// Seq2seq_9_88854283420255
// MI455X (gfx1250) — matrix operand order repaired and checked by reading, not compiled, not run on hardware
//
#include <hip/hip_runtime.h>
#include <hip/hip_bf16.h>
#include <math.h>

// ---------------- dims (fixed by the reference) ----------------
#define B    64
#define L    64
#define T    40
#define E    512
#define H    1024
#define DM   512
#define NH   8
#define DK   64
#define DV   64
#define V    30000
#define DIN  2050
#define DINP 2080              // DIN padded to multiple of 32 for WMMA K
#define G3H  3072              // 3*H
#define QKV  512               // NH*DK = NH*DV
#define OUTK 2048              // 2*H
#define NEGF (-1e9f)

typedef __attribute__((ext_vector_type(8)))  __bf16 v8bf;
typedef __attribute__((ext_vector_type(16))) __bf16 v16bf;
typedef __attribute__((ext_vector_type(8)))  float  v8f;

__device__ __forceinline__ float sigf(float x) { return 1.0f / (1.0f + __expf(-x)); }

// ---------------------------------------------------------------
// bf16 WMMA GEMM: C[M,N] = A[M,K] * Bt[N,K]^T (+bias[N])
// Register tile: 4 M-tiles x 1 N-tile per wave32 (M % 64 == 0).
// B (weight) fragment loaded ONCE per k-step, fed to 4 WMMAs.
// Load phase fully decoupled from compute phase so the scheduler
// can overlap WMMA with outstanding loads (partial loadcnt waits).
// Fragments use contiguous k = g*16 + e for both A and B; K multiple of 32.
// ---------------------------------------------------------------
__global__ __launch_bounds__(256)
void gemm_bf16_wmma(const __bf16* __restrict__ A, const __bf16* __restrict__ Bt,
                    float* __restrict__ C, int M, int N, int K,
                    const float* __restrict__ bias) {
    const int lane = threadIdx.x & 31;
    const int wave = threadIdx.x >> 5;
    const int tN   = N >> 4;
    const int work = (M >> 6) * tN;          // groups of 4 M-tiles
    const int w    = blockIdx.x * 8 + wave;
    if (w >= work) return;
    const int tmg = w / tN, tn = w % tN;
    const int g   = lane >> 4;               // half-wave selector
    const int r   = lane & 15;               // A-row / B-col within tile
    const __bf16* Ar[4];
#pragma unroll
    for (int i = 0; i < 4; ++i)
        Ar[i] = A + (size_t)(tmg * 64 + i * 16 + r) * K;
    const __bf16* Brow = Bt + (size_t)(tn * 16 + r) * K;
    v8f acc[4] = {};
    for (int k0 = 0; k0 < K; k0 += 32) {
        __builtin_prefetch(Brow + k0 + 512, 0, 3);   // global_prefetch_b8 (streamed weights)
        // ---- load phase: 1 B fragment + 4 A fragments, distinct regs ----
        // B 32x16 bf16 (N-major transposed weights): elems e -> K=k0+g*16+e
        v16bf b = *(const v16bf*)(Brow + k0 + g * 16);
        v16bf a[4];
#pragma unroll
        for (int i = 0; i < 4; ++i) {
            // A 16x32 bf16: lane r, elems e -> K=k0+g*16+e
            v8bf alo = *(const v8bf*)(Ar[i] + k0 + g * 16);
            v8bf ahi = *(const v8bf*)(Ar[i] + k0 + g * 16 + 8);
            a[i] = __builtin_shufflevector(alo, ahi,
                     0,1,2,3,4,5,6,7,8,9,10,11,12,13,14,15);
        }
        // ---- compute phase: 4 WMMAs share the B fragment ----
#pragma unroll
        for (int i = 0; i < 4; ++i)
            acc[i] = __builtin_amdgcn_wmma_f32_16x16x32_bf16(false, a[i], false, b,
                                                             (short)0, acc[i], false, false);
    }
    const int col = tn * 16 + r;
    const float bv = bias ? bias[col] : 0.0f;
#pragma unroll
    for (int i = 0; i < 4; ++i) {
#pragma unroll
        for (int ii = 0; ii < 8; ++ii) {
            const int row = tmg * 64 + i * 16 + g * 8 + ii;
            C[(size_t)row * N + col] = acc[i][ii] + bv;
        }
    }
}

// ---- weight convert + transpose: W[K,N] fp32 -> Wt[N,Kpad] bf16 (zero pad) ----
__global__ void convT_kernel(const float* __restrict__ W, __bf16* __restrict__ Wt,
                             int K, int N, int Kpad, long long total) {
    long long idx = (long long)blockIdx.x * 256 + threadIdx.x;
    if (idx >= total) return;
    int n = (int)(idx / Kpad), k = (int)(idx % Kpad);
    Wt[idx] = (k < K) ? (__bf16)W[(size_t)k * N + n] : (__bf16)0.0f;
}

// ---- xe = enc_emb[input_batch] -> bf16 (B*L rows of E) ----
__global__ void gather_embed(const float* __restrict__ emb, const int* __restrict__ ids,
                             __bf16* __restrict__ out, long long total) {
    long long idx = (long long)blockIdx.x * 256 + threadIdx.x;
    if (idx >= total) return;
    int row = (int)(idx / E), e = (int)(idx % E);
    out[idx] = (__bf16)emb[(size_t)ids[row] * E + e];
}

__global__ void build_fullseq(const int* __restrict__ target, int* __restrict__ fs) {
    int idx = blockIdx.x * 256 + threadIdx.x;
    if (idx >= B * (T + 1)) return;
    int b = idx / (T + 1), c = idx % (T + 1);
    fs[idx] = (c == 0) ? 1 : target[b * T + (c - 1)];   // SOS = 1
}

__global__ void zero_h(float* __restrict__ h, __bf16* __restrict__ hb) {
    int idx = blockIdx.x * 256 + threadIdx.x;
    if (idx >= B * H) return;
    h[idx] = 0.0f; hb[idx] = (__bf16)0.0f;
}

// ---- encoder GRU gate + length-mask freeze + enc_outs write ----
__global__ void enc_gate(const float* __restrict__ gx, const float* __restrict__ gh,
                         float* __restrict__ h, __bf16* __restrict__ hb,
                         float* __restrict__ eo, __bf16* __restrict__ eob,
                         const int* __restrict__ lens, int t) {
    int idx = blockIdx.x * 256 + threadIdx.x;      // B*H
    if (idx >= B * H) return;
    int b = idx >> 10, j = idx & 1023;
    const float* gxr = gx + (size_t)(b * L + t) * G3H;
    const float* ghr = gh + (size_t)b * G3H;
    float hp = h[idx];
    float rg = sigf(gxr[j]          + ghr[j]);
    float zg = sigf(gxr[H + j]      + ghr[H + j]);
    float ng = tanhf(gxr[2 * H + j] + rg * ghr[2 * H + j]);
    float hn = (1.0f - zg) * ng + zg * hp;
    bool  m  = t < lens[b];
    float hv = m ? hn : hp;
    h[idx] = hv; hb[idx] = (__bf16)hv;
    float ev = m ? hn : 0.0f;
    size_t o = (size_t)(b * L + t) * H + j;
    eo[o] = ev; eob[o] = (__bf16)ev;
}

// ---- decoder: kv = dec_emb[tok] + pos_emb[pos], bf16 (B*F rows of E) ----
__global__ void build_kv(const float* __restrict__ dec_emb, const float* __restrict__ pos_emb,
                         const int* __restrict__ fs, __bf16* __restrict__ kv, int t) {
    int idx = blockIdx.x * 256 + threadIdx.x;      // B*8*E
    if (idx >= B * 8 * E) return;
    int r = idx >> 9, e = idx & 511;
    int b = r >> 3, f = r & 7;
    int start = t - 7; if (start < 0) start = 0;
    int tok = fs[b * (T + 1) + start + f];
    int pos = start + f + 1;
    kv[idx] = (__bf16)(dec_emb[(size_t)tok * E + e] + pos_emb[(size_t)pos * E + e]);
}

// ---- windowed MHA: one block(64) per (b,head) ----
__global__ __launch_bounds__(64)
void self_attn(const float* __restrict__ q, const float* __restrict__ ks,
               const float* __restrict__ vs, __bf16* __restrict__ ctx_in, int t) {
    int b = blockIdx.x >> 3, hh = blockIdx.x & 7;
    int tid = threadIdx.x;
    __shared__ float qs[64];
    __shared__ float p[8];
    qs[tid] = q[(size_t)b * QKV + hh * DK + tid];
    __syncthreads();
    int start = t - 7; if (start < 0) start = 0;
    if (tid < 8) {
        const float* kr = ks + (size_t)(b * 8 + tid) * QKV + hh * DK;
        float acc = 0.0f;
        for (int d = 0; d < DK; ++d) acc += qs[d] * kr[d];
        bool valid = (start + tid) < (t + 1);
        p[tid] = valid ? acc * 0.125f : NEGF;      // 1/sqrt(64)
    }
    __syncthreads();
    if (tid == 0) {
        float mx = p[0];
        for (int f = 1; f < 8; ++f) mx = fmaxf(mx, p[f]);
        float s = 0.0f;
        for (int f = 0; f < 8; ++f) { p[f] = __expf(p[f] - mx); s += p[f]; }
        float inv = 1.0f / s;
        for (int f = 0; f < 8; ++f) p[f] *= inv;
    }
    __syncthreads();
    float acc = 0.0f;
    for (int f = 0; f < 8; ++f)
        acc += p[f] * vs[(size_t)(b * 8 + f) * QKV + hh * DV + tid];
    ctx_in[(size_t)b * QKV + hh * DV + tid] = (__bf16)acc;
}

// ---- additive attention scores e[b,l] = v_a . tanh(hWq[b]+enc_proj[b,l]) ----
__global__ __launch_bounds__(256)
void attn_scores(const float* __restrict__ hWq, const float* __restrict__ enc_proj,
                 const float* __restrict__ v_a, float* __restrict__ e_s) {
    int bl = blockIdx.x, b = bl >> 6;
    __shared__ float red[256];
    float acc = 0.0f;
    for (int j = threadIdx.x; j < H; j += 256)
        acc += tanhf(hWq[(size_t)b * H + j] + enc_proj[(size_t)bl * H + j]) * v_a[j];
    red[threadIdx.x] = acc; __syncthreads();
    for (int s = 128; s > 0; s >>= 1) {
        if (threadIdx.x < s) red[threadIdx.x] += red[threadIdx.x + s];
        __syncthreads();
    }
    if (threadIdx.x == 0) e_s[bl] = red[0];
}

__global__ __launch_bounds__(64)
void softmax_enc(const float* __restrict__ e_s, const int* __restrict__ lens,
                 float* __restrict__ p_enc) {
    int b = blockIdx.x, l = threadIdx.x;
    __shared__ float s[64];
    __shared__ float mx, sum;
    float v = (l < lens[b]) ? e_s[b * L + l] : NEGF;
    s[l] = v; __syncthreads();
    if (l == 0) { float m = s[0]; for (int i = 1; i < L; ++i) m = fmaxf(m, s[i]); mx = m; }
    __syncthreads();
    float ex = __expf(v - mx);
    s[l] = ex; __syncthreads();
    if (l == 0) { float q = 0.0f; for (int i = 0; i < L; ++i) q += s[i]; sum = q; }
    __syncthreads();
    p_enc[b * L + l] = ex / sum;
}

__global__ __launch_bounds__(256)
void ctx_enc_kernel(const float* __restrict__ p_enc, const float* __restrict__ eo,
                    float* __restrict__ ctx) {
    int b = blockIdx.x;
    __shared__ float p[64];
    if (threadIdx.x < 64) p[threadIdx.x] = p_enc[b * L + threadIdx.x];
    __syncthreads();
    for (int j = threadIdx.x; j < H; j += 256) {
        float acc = 0.0f;
        for (int l = 0; l < L; ++l) acc += p[l] * eo[(size_t)(b * L + l) * H + j];
        ctx[b * H + j] = acc;
    }
}

// ---- x_d = [dec_emb[y_prev] | ctx_enc | ctx_self | i,j | pad] -> bf16 ----
__global__ void assemble_xd(const float* __restrict__ dec_emb, const int* __restrict__ fs,
                            const float* __restrict__ ctx_enc, const float* __restrict__ ctx_self,
                            __bf16* __restrict__ xd, int t) {
    int idx = blockIdx.x * 256 + threadIdx.x;      // B*DINP
    if (idx >= B * DINP) return;
    int b = idx / DINP, c = idx % DINP;
    float v;
    if      (c < E)          { int y = fs[b * (T + 1) + t]; v = dec_emb[(size_t)y * E + c]; }
    else if (c < E + H)        v = ctx_enc[b * H + (c - E)];
    else if (c < E + H + DM)   v = ctx_self[b * DM + (c - E - H)];
    else if (c == DIN - 2)     v = (float)(t / 8);
    else if (c == DIN - 1)     v = (float)(t % 8);
    else                       v = 0.0f;
    xd[idx] = (__bf16)v;
}

// ---- decoder GRU gate; also packs out_in = [h_new | ctx_enc] bf16 ----
__global__ void dec_gate(const float* __restrict__ gx, const float* __restrict__ gh,
                         float* __restrict__ h, __bf16* __restrict__ hb,
                         const float* __restrict__ ctx_enc, __bf16* __restrict__ out_in) {
    int idx = blockIdx.x * 256 + threadIdx.x;      // B*H
    if (idx >= B * H) return;
    int b = idx >> 10, j = idx & 1023;
    const float* gxr = gx + (size_t)b * G3H;
    const float* ghr = gh + (size_t)b * G3H;
    float hp = h[idx];
    float rg = sigf(gxr[j]          + ghr[j]);
    float zg = sigf(gxr[H + j]      + ghr[H + j]);
    float ng = tanhf(gxr[2 * H + j] + rg * ghr[2 * H + j]);
    float hn = (1.0f - zg) * ng + zg * hp;
    h[idx] = hn; hb[idx] = (__bf16)hn;
    out_in[(size_t)b * OUTK + j]     = (__bf16)hn;
    out_in[(size_t)b * OUTK + H + j] = (__bf16)ctx_enc[b * H + j];
}

// ---- per-row log-softmax NLL over V=30000 ----
__global__ __launch_bounds__(256)
void loss_rows(const float* __restrict__ logits, const int* __restrict__ target,
               float* __restrict__ loss_b, int t) {
    int b = blockIdx.x;
    const float* row = logits + (size_t)b * V;
    __shared__ float red[256];
    float mx = -3.4e38f;
    for (int j = threadIdx.x; j < V; j += 256) mx = fmaxf(mx, row[j]);
    red[threadIdx.x] = mx; __syncthreads();
    for (int s = 128; s > 0; s >>= 1) {
        if (threadIdx.x < s) red[threadIdx.x] = fmaxf(red[threadIdx.x], red[threadIdx.x + s]);
        __syncthreads();
    }
    mx = red[0]; __syncthreads();
    float sum = 0.0f;
    for (int j = threadIdx.x; j < V; j += 256) sum += __expf(row[j] - mx);
    red[threadIdx.x] = sum; __syncthreads();
    for (int s = 128; s > 0; s >>= 1) {
        if (threadIdx.x < s) red[threadIdx.x] += red[threadIdx.x + s];
        __syncthreads();
    }
    if (threadIdx.x == 0) {
        int tgt = target[b * T + t];
        float logp = (row[tgt] - mx) - logf(red[0]);
        loss_b[t * B + b] = -logp * (1.0f / (float)B);
    }
}

__global__ __launch_bounds__(256)
void sum_loss(const float* __restrict__ loss_b, float* __restrict__ out) {
    __shared__ float red[256];
    float acc = 0.0f;
    for (int i = threadIdx.x; i < T * B; i += 256) acc += loss_b[i];
    red[threadIdx.x] = acc; __syncthreads();
    for (int s = 128; s > 0; s >>= 1) {
        if (threadIdx.x < s) red[threadIdx.x] += red[threadIdx.x + s];
        __syncthreads();
    }
    if (threadIdx.x == 0) out[0] = red[0];
}

// ---------------------------------------------------------------
extern "C" void kernel_launch(void* const* d_in, const int* in_sizes, int n_in,
                              void* d_out, int out_size, void* d_ws, size_t ws_size,
                              hipStream_t stream) {
    (void)in_sizes; (void)n_in; (void)out_size; (void)ws_size;
    const int*   input_batch   = (const int*)d_in[0];
    const int*   input_lengths = (const int*)d_in[1];
    const int*   target_batch  = (const int*)d_in[2];
    const float* enc_emb = (const float*)d_in[3];
    const float* dec_emb = (const float*)d_in[4];
    const float* pos_emb = (const float*)d_in[5];
    const float* Wx_e = (const float*)d_in[6];
    const float* Wh_e = (const float*)d_in[7];
    const float* b_e  = (const float*)d_in[8];
    const float* Wx_d = (const float*)d_in[9];
    const float* Wh_d = (const float*)d_in[10];
    const float* b_d  = (const float*)d_in[11];
    const float* Wq_a = (const float*)d_in[12];
    const float* Wk_a = (const float*)d_in[13];
    const float* v_a  = (const float*)d_in[14];
    const float* Wqh  = (const float*)d_in[15];
    const float* Wks  = (const float*)d_in[16];
    const float* Wvs  = (const float*)d_in[17];
    const float* Wo_s = (const float*)d_in[18];
    const float* W_out = (const float*)d_in[19];
    const float* b_out = (const float*)d_in[20];
    float* out = (float*)d_out;

    // ---- carve workspace (256B aligned) ----
    char* p = (char*)d_ws;
    auto carve = [&p](size_t bytes) {
        char* r = p;
        p += (bytes + 255) & ~(size_t)255;
        return (void*)r;
    };
    __bf16* wt_xe  = (__bf16*)carve((size_t)G3H * E * 2);
    __bf16* wt_he  = (__bf16*)carve((size_t)G3H * H * 2);
    __bf16* wt_qa  = (__bf16*)carve((size_t)H * H * 2);
    __bf16* wt_ka  = (__bf16*)carve((size_t)H * H * 2);
    __bf16* wt_qh  = (__bf16*)carve((size_t)QKV * H * 2);
    __bf16* wt_ks  = (__bf16*)carve((size_t)QKV * E * 2);
    __bf16* wt_vs  = (__bf16*)carve((size_t)QKV * E * 2);
    __bf16* wt_os  = (__bf16*)carve((size_t)DM * QKV * 2);
    __bf16* wt_xd  = (__bf16*)carve((size_t)G3H * DINP * 2);
    __bf16* wt_hd  = (__bf16*)carve((size_t)G3H * H * 2);
    __bf16* wt_out = (__bf16*)carve((size_t)V * OUTK * 2);
    __bf16* xe     = (__bf16*)carve((size_t)B * L * E * 2);
    float*  gx_e   = (float*) carve((size_t)B * L * G3H * 4);
    float*  h      = (float*) carve((size_t)B * H * 4);
    __bf16* hb     = (__bf16*)carve((size_t)B * H * 2);
    float*  gh     = (float*) carve((size_t)B * G3H * 4);
    float*  eo     = (float*) carve((size_t)B * L * H * 4);
    __bf16* eob    = (__bf16*)carve((size_t)B * L * H * 2);
    float*  eproj  = (float*) carve((size_t)B * L * H * 4);
    int*    fs     = (int*)   carve((size_t)B * (T + 1) * 4);
    __bf16* kv     = (__bf16*)carve((size_t)B * 8 * E * 2);
    float*  q      = (float*) carve((size_t)B * QKV * 4);
    float*  ks     = (float*) carve((size_t)B * 8 * QKV * 4);
    float*  vs     = (float*) carve((size_t)B * 8 * QKV * 4);
    __bf16* ctxsi  = (__bf16*)carve((size_t)B * QKV * 2);
    float*  ctxs   = (float*) carve((size_t)B * DM * 4);
    float*  hWq    = (float*) carve((size_t)B * H * 4);
    float*  e_s    = (float*) carve((size_t)B * L * 4);
    float*  p_enc  = (float*) carve((size_t)B * L * 4);
    float*  ctxe   = (float*) carve((size_t)B * H * 4);
    __bf16* xd     = (__bf16*)carve((size_t)B * DINP * 2);
    float*  gx_d   = (float*) carve((size_t)B * G3H * 4);
    __bf16* outin  = (__bf16*)carve((size_t)B * OUTK * 2);
    float*  logits = (float*) carve((size_t)B * V * 4);
    float*  lossb  = (float*) carve((size_t)T * B * 4);

    auto conv = [&](const float* W, __bf16* Wt, int K, int N, int Kpad) {
        long long total = (long long)N * Kpad;
        convT_kernel<<<(unsigned)((total + 255) / 256), 256, 0, stream>>>(W, Wt, K, N, Kpad, total);
    };
    auto gemm = [&](const __bf16* A, const __bf16* Bt, float* C, int M, int N, int K,
                    const float* bias) {
        int work = (M / 64) * (N / 16);      // 4 M-tiles per wave
        gemm_bf16_wmma<<<(work + 7) / 8, 256, 0, stream>>>(A, Bt, C, M, N, K, bias);
    };

    // ---- one-time setup (per launch; deterministic) ----
    conv(Wx_e, wt_xe, E,   G3H, E);
    conv(Wh_e, wt_he, H,   G3H, H);
    conv(Wq_a, wt_qa, H,   H,   H);
    conv(Wk_a, wt_ka, H,   H,   H);
    conv(Wqh,  wt_qh, H,   QKV, H);
    conv(Wks,  wt_ks, E,   QKV, E);
    conv(Wvs,  wt_vs, E,   QKV, E);
    conv(Wo_s, wt_os, QKV, DM,  QKV);
    conv(Wx_d, wt_xd, DIN, G3H, DINP);
    conv(Wh_d, wt_hd, H,   G3H, H);
    conv(W_out, wt_out, OUTK, V, OUTK);

    {
        long long total = (long long)B * L * E;
        gather_embed<<<(unsigned)((total + 255) / 256), 256, 0, stream>>>(enc_emb, input_batch, xe, total);
    }
    build_fullseq<<<(B * (T + 1) + 255) / 256, 256, 0, stream>>>(target_batch, fs);
    zero_h<<<(B * H + 255) / 256, 256, 0, stream>>>(h, hb);

    // encoder input GEMM for all timesteps at once
    gemm(xe, wt_xe, gx_e, B * L, G3H, E, b_e);

    // ---- encoder recurrence ----
    for (int t = 0; t < L; ++t) {
        gemm(hb, wt_he, gh, B, G3H, H, nullptr);
        enc_gate<<<(B * H + 255) / 256, 256, 0, stream>>>(gx_e, gh, h, hb, eo, eob, input_lengths, t);
    }
    // attention keys for all encoder positions
    gemm(eob, wt_ka, eproj, B * L, H, H, nullptr);

    // ---- decoder ----
    for (int t = 0; t < T; ++t) {
        build_kv<<<(B * 8 * E + 255) / 256, 256, 0, stream>>>(dec_emb, pos_emb, fs, kv, t);
        gemm(hb, wt_qh, q,  B,     QKV, H, nullptr);
        gemm(kv, wt_ks, ks, B * 8, QKV, E, nullptr);
        gemm(kv, wt_vs, vs, B * 8, QKV, E, nullptr);
        self_attn<<<B * NH, 64, 0, stream>>>(q, ks, vs, ctxsi, t);
        gemm(ctxsi, wt_os, ctxs, B, DM, QKV, nullptr);
        gemm(hb, wt_qa, hWq, B, H, H, nullptr);
        attn_scores<<<B * L, 256, 0, stream>>>(hWq, eproj, v_a, e_s);
        softmax_enc<<<B, 64, 0, stream>>>(e_s, input_lengths, p_enc);
        ctx_enc_kernel<<<B, 256, 0, stream>>>(p_enc, eo, ctxe);
        assemble_xd<<<(B * DINP + 255) / 256, 256, 0, stream>>>(dec_emb, fs, ctxe, ctxs, xd, t);
        gemm(xd, wt_xd, gx_d, B, G3H, DINP, b_d);
        gemm(hb, wt_hd, gh,   B, G3H, H, nullptr);
        dec_gate<<<(B * H + 255) / 256, 256, 0, stream>>>(gx_d, gh, h, hb, ctxe, outin);
        gemm(outin, wt_out, logits, B, V, OUTK, b_out);
        loss_rows<<<B, 256, 0, stream>>>(logits, target_batch, lossb, t);
    }
    sum_loss<<<1, 256, 0, stream>>>(lossb, out);
}
